// EnhancedBioKANModel_37538014167115
// MI455X (gfx1250) — matrix operand order repaired and checked by reading, not compiled, not run on hardware
//
#include <hip/hip_runtime.h>

typedef __attribute__((ext_vector_type(16))) __bf16 v16bf;
typedef __attribute__((ext_vector_type(8)))  float  v8f;
typedef unsigned short u16;

#if __has_builtin(__builtin_amdgcn_global_load_async_to_lds_b128)
#define ASYNC_LDS 1
#else
#define ASYNC_LDS 0
#endif

// GCC-style vector (matches builtin's parameter type per hipcc diagnostic)
typedef int v4i_g __attribute__((vector_size(16)));
typedef __attribute__((address_space(1))) v4i_g* gas_b128;
typedef __attribute__((address_space(3))) v4i_g* las_b128;

// ---------------------------------------------------------------- helpers
__device__ __forceinline__ u16 f2bf(float f) {
    unsigned u = __builtin_bit_cast(unsigned, f);
    unsigned r = u + 0x7FFFu + ((u >> 16) & 1u);       // round-to-nearest-even
    if ((u & 0x7FFFFFFFu) > 0x7F800000u) r = u | 0x00400000u; // quiet NaN
    return (u16)(r >> 16);
}

__device__ __forceinline__ void copy16(const u16* g, u16* l) {
#if ASYNC_LDS
    __builtin_amdgcn_global_load_async_to_lds_b128(
        (gas_b128)(u16*)g, (las_b128)l, 0, 0);
#else
    *(uint4*)l = *(const uint4*)g;
#endif
}

__device__ __forceinline__ void async_wait_all() {
#if ASYNC_LDS
#if __has_builtin(__builtin_amdgcn_s_wait_asynccnt)
    __builtin_amdgcn_s_wait_asynccnt(0);
#else
    asm volatile("s_wait_asynccnt 0x0" ::: "memory");
#endif
#endif
}

__global__ void cvt_f32_to_bf16(const float* __restrict__ src,
                                u16* __restrict__ dst, long long n) {
    long long i = ((long long)blockIdx.x * blockDim.x + threadIdx.x) * 4;
    if (i >= n) return;
    float4 f = *(const float4*)(src + i);
    uint2 o;
    o.x = (unsigned)f2bf(f.x) | ((unsigned)f2bf(f.y) << 16);
    o.y = (unsigned)f2bf(f.z) | ((unsigned)f2bf(f.w) << 16);
    *(uint2*)(dst + i) = o;
}

// ---------------------------------------------------------------- GEMM
// C[M,N] = A[M,K] (bf16, row-major) * B[N,K]^T (bf16, row-major) + bias[N]
// EPI: 0 = f32 out; 1 = relu, f32 + bf16 out; 2 = bf16 out; 3 = +Res, f32 + bf16 out
union FragBF { v16bf v; uint4 u[2]; };

template <int EPI>
__global__ void __launch_bounds__(256)
gemm_bf16(const u16* __restrict__ A, const u16* __restrict__ B,
          const float* __restrict__ bias, float* __restrict__ Cf,
          u16* __restrict__ Cb, const float* __restrict__ Res,
          int M, int N, int K) {
    constexpr int BK = 64, PS = 72;          // PS: padded LDS row stride (elems)
    __shared__ u16 As[2][128 * PS];
    __shared__ u16 Bs[2][128 * PS];

    const int tid  = threadIdx.x;
    const int lane = tid & 31, wid = tid >> 5;
    const int wm = wid & 1, wn = wid >> 1;   // 2 x 4 wave grid -> 64 x 32 per wave
    const int lm = lane & 15, lh = lane >> 4;
    const int bm0 = blockIdx.y * 128;
    const int bn0 = blockIdx.x * 128;
    const bool fullN = (bn0 + 128 <= N);

    v8f acc[4][2] = {};

    // global->LDS: 128 rows x 8 chunks(16B); 256 thr -> 4 row-passes per thread
    const int rowp  = tid >> 3;              // 0..31
    const int chunk = (tid & 7) * 8;         // element offset of 16B chunk

    auto loadA = [&](int buf, int k0) {
#pragma unroll
        for (int p = 0; p < 4; p++) {
            const int r = rowp + p * 32;
            copy16(A + (size_t)(bm0 + r) * K + k0 + chunk, &As[buf][r * PS + chunk]);
        }
    };
    auto loadB = [&](int buf, int k0) {
#pragma unroll
        for (int p = 0; p < 4; p++) {
            const int r = rowp + p * 32;
            u16* dst = &Bs[buf][r * PS + chunk];
            const u16* src = B + (size_t)(bn0 + r) * K + k0 + chunk;
            if (fullN) {
                copy16(src, dst);
            } else {                          // tail column (Wout): guarded sync path
                uint4 val = make_uint4(0u, 0u, 0u, 0u);
                if (bn0 + r < N) val = *(const uint4*)src;
                *(uint4*)dst = val;
            }
        }
    };
    auto mma = [&](int buf) {
#pragma unroll
        for (int ksub = 0; ksub < 2; ksub++) {
            const int kb = ksub * 32;
            FragBF af[4], bf_[2];
#pragma unroll
            for (int i = 0; i < 4; i++) {
                // ISA 16-bit A(16x32): lanes0-15 K=0..7/16..23, lanes16-31 K=8..15/24..31
                const u16* p = &As[buf][(wm * 64 + i * 16 + lm) * PS + kb + lh * 8];
                af[i].u[0] = *(const uint4*)p;
                af[i].u[1] = *(const uint4*)(p + 16);
            }
#pragma unroll
            for (int j = 0; j < 2; j++) {
                // B(32x16), same k order as A: lanes0-15 K=0..7/16..23, lanes16-31 K=8..15/24..31
                const u16* p = &Bs[buf][(wn * 32 + j * 16 + lm) * PS + kb + lh * 8];
                bf_[j].u[0] = *(const uint4*)p;
                bf_[j].u[1] = *(const uint4*)(p + 16);
            }
#pragma unroll
            for (int i = 0; i < 4; i++)
#pragma unroll
                for (int j = 0; j < 2; j++)
                    acc[i][j] = __builtin_amdgcn_wmma_f32_16x16x32_bf16(
                        false, af[i].v, false, bf_[j].v, (short)0, acc[i][j],
                        false, false);
        }
    };

    loadA(0, 0);
    loadB(0, 0);
    const int nk = K / BK;
    for (int ks = 0; ks < nk; ks++) {
        async_wait_all();                    // our async writes into buf are done
        __syncthreads();                     // publish buf to all waves
        const int buf = ks & 1;
        if (ks + 1 < nk) { loadA(buf ^ 1, (ks + 1) * BK); loadB(buf ^ 1, (ks + 1) * BK); }
        mma(buf);
    }

    // epilogue: C/D layout -> VGPR r, lanes0-15: M=r, lanes16-31: M=r+8; N=lane&15
#pragma unroll
    for (int i = 0; i < 4; i++) {
        const int mrow0 = bm0 + wm * 64 + i * 16 + lh * 8;
#pragma unroll
        for (int j = 0; j < 2; j++) {
            const int n = bn0 + wn * 32 + j * 16 + lm;
            if (n < N) {
                const float bv = bias[n];
#pragma unroll
                for (int r = 0; r < 8; r++) {
                    const size_t idx = (size_t)(mrow0 + r) * N + n;
                    float val = acc[i][j][r] + bv;
                    if (EPI == 1) val = fmaxf(val, 0.f);
                    if (EPI == 3) val += Res[idx];
                    if (EPI == 0 || EPI == 1 || EPI == 3) Cf[idx] = val;
                    if (EPI == 1 || EPI == 2 || EPI == 3) Cb[idx] = f2bf(val);
                }
            }
        }
    }
}

// ---------------------------------------------------------------- LayerNorm
// RELU: zB = relu(LN(T)*g + b) as bf16.   RESID: hF += LN(T)*g + b, hB = bf16(hF)
template <bool RELU, bool RESID>
__global__ void __launch_bounds__(256)
ln_rows(const float* __restrict__ T, const float* __restrict__ gw,
        const float* __restrict__ bw, float* __restrict__ hF,
        u16* __restrict__ hB, u16* __restrict__ zB, int R) {
    __shared__ float rs[256], rs2[256];
    const int row = blockIdx.x, tid = threadIdx.x;
    const float* xr = T + (size_t)row * R;
    const int per = R >> 10;                  // float4's per thread (R=2048->2, 4096->4)
    float4 v[4];
    float s = 0.f, s2 = 0.f;
    for (int i = 0; i < per; i++) {
        v[i] = *(const float4*)(xr + (size_t)(tid + i * 256) * 4);
        s  += v[i].x + v[i].y + v[i].z + v[i].w;
        s2 += v[i].x * v[i].x + v[i].y * v[i].y + v[i].z * v[i].z + v[i].w * v[i].w;
    }
    rs[tid] = s; rs2[tid] = s2;
    __syncthreads();
    for (int off = 128; off > 0; off >>= 1) {
        if (tid < off) { rs[tid] += rs[tid + off]; rs2[tid] += rs2[tid + off]; }
        __syncthreads();
    }
    const float inv  = 1.f / (float)R;
    const float mean = rs[0] * inv;
    const float var  = rs2[0] * inv - mean * mean;
    const float rstd = rsqrtf(var + 1e-5f);
    for (int i = 0; i < per; i++) {
        const int c = (tid + i * 256) * 4;
        const float4 g4 = *(const float4*)(gw + c);
        const float4 b4 = *(const float4*)(bw + c);
        float o[4] = { (v[i].x - mean) * rstd * g4.x + b4.x,
                       (v[i].y - mean) * rstd * g4.y + b4.y,
                       (v[i].z - mean) * rstd * g4.z + b4.z,
                       (v[i].w - mean) * rstd * g4.w + b4.w };
#pragma unroll
        for (int k = 0; k < 4; k++) {
            const size_t idx = (size_t)row * R + c + k;
            float ov = o[k];
            if (RELU) ov = fmaxf(ov, 0.f);
            if (RESID) {
                float nh = hF[idx] + ov;
                hF[idx] = nh;
                hB[idx] = f2bf(nh);
            } else {
                zB[idx] = f2bf(ov);
            }
        }
    }
}

// ---------------------------------------------------------------- host
extern "C" void kernel_launch(void* const* d_in, const int* in_sizes, int n_in,
                              void* d_out, int out_size, void* d_ws, size_t ws_size,
                              hipStream_t stream) {
    const int Bsz = 4096, IN = 3072, D = 2048, D2 = 4096, NC = 1000, NB = 3;

    const float* x    = (const float*)d_in[0];
    const float* Wi   = (const float*)d_in[1];
    const float* bi   = (const float*)d_in[2];
    const float* Wv   = (const float*)d_in[7];
    const float* bv   = (const float*)d_in[8];
    const float* Wo   = (const float*)d_in[9];
    const float* bo   = (const float*)d_in[10];
    const float* W1   = (const float*)d_in[13];
    const float* b1   = (const float*)d_in[14];
    const float* g1   = (const float*)d_in[15];
    const float* be1  = (const float*)d_in[16];
    const float* W2   = (const float*)d_in[17];
    const float* b2   = (const float*)d_in[18];
    const float* g2   = (const float*)d_in[19];
    const float* be2  = (const float*)d_in[20];
    const float* Wout = (const float*)d_in[21];
    const float* bout = (const float*)d_in[22];
    // Wq/bq/Wk/bk/conv_w/conv_b are provably dead (softmax over S=1 is 1.0)

    char* p = (char*)d_ws;
    auto take = [&](size_t bytes) { char* r = p; p += ((bytes + 255) / 256) * 256; return r; };
    u16*   xb = (u16*)  take((size_t)Bsz * IN * 2);
    u16*   wb = (u16*)  take((size_t)D2 * D * 2);     // per-GEMM bf16 weight slot
    float* hF = (float*)take((size_t)Bsz * D * 4);
    u16*   hB = (u16*)  take((size_t)Bsz * D * 2);
    float* tF = (float*)take((size_t)Bsz * D2 * 4);   // also reused for [B,D] tmp
    u16*   zB = (u16*)  take((size_t)Bsz * D2 * 2);
    u16*   vB = (u16*)  take((size_t)Bsz * D * 2);

    auto cvt = [&](const float* s, u16* d, long long n) {
        cvt_f32_to_bf16<<<(unsigned)((n / 4 + 255) / 256), 256, 0, stream>>>(s, d, n);
    };

    // h = relu(x @ Wi^T + bi)
    cvt(x, xb, (long long)Bsz * IN);
    cvt(Wi, wb, (long long)D * IN);
    gemm_bf16<1><<<dim3(D / 128, Bsz / 128), 256, 0, stream>>>(
        xb, wb, bi, hF, hB, nullptr, Bsz, D, IN);

    for (int l = 0; l < NB; l++) {
        // attention collapses to: h += (h @ Wv^T + bv) @ Wo^T + bo
        cvt(Wv + (size_t)l * D * D, wb, (long long)D * D);
        gemm_bf16<2><<<dim3(D / 128, Bsz / 128), 256, 0, stream>>>(
            hB, wb, bv + (size_t)l * D, nullptr, vB, nullptr, Bsz, D, D);
        cvt(Wo + (size_t)l * D * D, wb, (long long)D * D);
        gemm_bf16<3><<<dim3(D / 128, Bsz / 128), 256, 0, stream>>>(
            vB, wb, bo + (size_t)l * D, hF, hB, hF, Bsz, D, D);
        // z = relu(LN(h @ W1^T + b1))
        cvt(W1 + (size_t)l * D2 * D, wb, (long long)D2 * D);
        gemm_bf16<0><<<dim3(D2 / 128, Bsz / 128), 256, 0, stream>>>(
            hB, wb, b1 + (size_t)l * D2, tF, nullptr, nullptr, Bsz, D2, D);
        ln_rows<true, false><<<Bsz, 256, 0, stream>>>(
            tF, g1 + (size_t)l * D2, be1 + (size_t)l * D2, nullptr, nullptr, zB, D2);
        // h += LN(z @ W2^T + b2)
        cvt(W2 + (size_t)l * D * D2, wb, (long long)D * D2);
        gemm_bf16<0><<<dim3(D / 128, Bsz / 128), 256, 0, stream>>>(
            zB, wb, b2 + (size_t)l * D, tF, nullptr, nullptr, Bsz, D, D2);
        ln_rows<false, true><<<Bsz, 256, 0, stream>>>(
            tF, g2 + (size_t)l * D, be2 + (size_t)l * D, hF, hB, nullptr, D);
    }

    // out = h @ Wout^T + bout   (N=1000, guarded tiles)
    cvt(Wout, wb, (long long)NC * D);
    gemm_bf16<0><<<dim3((NC + 127) / 128, Bsz / 128), 256, 0, stream>>>(
        hB, wb, bout, (float*)d_out, nullptr, nullptr, Bsz, NC, D);

    (void)in_sizes; (void)n_in; (void)out_size; (void)ws_size;
}
